// Head_76759655514779
// MI455X (gfx1250) — matrix operand order repaired and checked by reading, not compiled, not run on hardware
//
#include <hip/hip_runtime.h>
#include <math.h>

// ---------------------------------------------------------------------------
// Hybrid attention head for MI455X (gfx1250, wave32, WMMA).
// All heavy math runs through one bf16 NT-GEMM kernel built around
// v_wmma_f32_16x16x32_bf16 with double-buffered LDS staging fed by
// GLOBAL_LOAD_ASYNC_TO_LDS_B128 (ASYNCcnt / s_wait_asynccnt) when available.
// Pipeline:
//   cvt(x,W*) -> q,k GEMM -> v GEMM (stores v^T) -> scores GEMM (fused
//   scale/causal/relu/decay) -> softmax+entropy -> flag -> [latent: 2 GEMMs +
//   softmax | quant: top-16 kernel] -> out GEMM (attn @ v via v^T).
// Branch select is device-side (flag in workspace), graph-capture safe.
// Workspace: ~421 MB.
// ---------------------------------------------------------------------------

typedef __bf16 bf16;
typedef __attribute__((ext_vector_type(8)))  __bf16 v8bf;
typedef __attribute__((ext_vector_type(16))) __bf16 v16bf;
typedef __attribute__((ext_vector_type(8)))  float  v8f;

constexpr int Bn  = 16;
constexpr int Tn  = 1024;
constexpr int Dn  = 1024;
constexpr int HSn = 2048;
constexpr int Ln  = 1024;
constexpr int BTn = Bn * Tn;

// GEMM block tiling
constexpr int BM = 128, BN = 128, BK = 32;
constexpr int LROW = 40;            // bf16 elems per LDS row (32 + 8 pad = 80B)
constexpr int BOFF = BM * LROW;     // B-tile base inside an LDS buffer

#if defined(__gfx1250__) && \
    __has_builtin(__builtin_amdgcn_global_load_async_to_lds_b128) && \
    __has_builtin(__builtin_amdgcn_s_wait_asynccnt)
#define USE_ASYNC_LDS 1
// Builtin signature (from hipcc diagnostic): (v4i AS1*, v4i AS3*, imm, imm)
typedef int vi4 __attribute__((vector_size(16)));
typedef __attribute__((address_space(1))) vi4* gv4_ptr;
typedef __attribute__((address_space(3))) vi4* lv4_ptr;
#define ASG(p) ((gv4_ptr)(void*)(p))
#define ASL(p) ((lv4_ptr)(void*)(p))
#else
#define USE_ASYNC_LDS 0
#endif

__device__ __forceinline__ bf16 tobf(float f) {
  unsigned u = __builtin_bit_cast(unsigned, f);
  u += 0x7FFFu + ((u >> 16) & 1u);                 // round-to-nearest-even
  unsigned short h = (unsigned short)(u >> 16);
  return __builtin_bit_cast(bf16, h);
}

__global__ __launch_bounds__(256)
void cvt_bf16(const float* __restrict__ in, bf16* __restrict__ out, int n) {
  int i = blockIdx.x * 256 + threadIdx.x;
  if (i < n) out[i] = tobf(in[i]);
}

// ---------------------------------------------------------------------------
// NT GEMM: C[M,N] = A[M,K] * Bt[N,K]^T, bf16 in / fp32 accumulate.
// Block: 256 threads = 8 waves as 4(M) x 2(N); wave tile 32x64 (8 WMMA accs).
// Block tile 128x128x32, LDS double buffered (40KB), async-to-LDS staging.
// Grid: (N/128, M/128, batch).  EPI: 0=bf16, 1=bf16 per-batch-transposed
// (v^T), 2=scores(scale,causal,relu,decay)->f32, 3=relu->bf16, 4=f32.
// GUARD: 0 always, 1 run iff *flag!=0, 2 run iff *flag==0.
// ---------------------------------------------------------------------------
template <int EPI, int GUARD>
__global__ __launch_bounds__(256)
void gemm_bf16_nt(const bf16* __restrict__ A, long long sA, int lda,
                  const bf16* __restrict__ Bt, long long sB, int ldb,
                  void* __restrict__ Cv, long long sC, int ldc,
                  int K, float scale, const int* __restrict__ flag)
{
  if constexpr (GUARD == 1) { if (*flag == 0) return; }
  if constexpr (GUARD == 2) { if (*flag != 0) return; }

  __shared__ __align__(16) bf16 sh[2][(BM + BN) * LROW];   // 40 KB

  const int z = blockIdx.z;
  A  += (long long)z * sA;
  Bt += (long long)z * sB;

  const int tid  = threadIdx.x;
  const int lane = tid & 31;
  const int wave = tid >> 5;
  const int wm = wave >> 1;                // 4 waves in M
  const int wn = wave & 1;                 // 2 waves in N
  const int bm0 = blockIdx.y * BM;
  const int bn0 = blockIdx.x * BN;
  const int m0w = wm * 32;                 // wave row base inside block tile
  const int n0w = wn * 64;                 // wave col base inside block tile

  const int lr = lane & 15;                // row/col within 16
  const int lh = lane >> 4;                // half-wave select
  const int aK = lh * 8;                   // A: K chunks {aK, aK+16}
  const int bK = lh * 8;                   // B: K chunks {bK, bK+16}

  // staging assignment: thread -> (tile row, 16-elem half of the K-slice)
  const int srow = tid >> 1;
  const int sh16 = (tid & 1) * 16;

  auto stage = [&](int buf, int kk) {
    const bf16* ga = A  + (long long)(bm0 + srow) * lda + (kk + sh16);
    const bf16* gb = Bt + (long long)(bn0 + srow) * ldb + (kk + sh16);
    bf16* la = &sh[buf][srow * LROW + sh16];
    bf16* lb = &sh[buf][BOFF + srow * LROW + sh16];
#if USE_ASYNC_LDS
    __builtin_amdgcn_global_load_async_to_lds_b128(ASG(ga),     ASL(la),     0, 0);
    __builtin_amdgcn_global_load_async_to_lds_b128(ASG(ga + 8), ASL(la + 8), 0, 0);
    __builtin_amdgcn_global_load_async_to_lds_b128(ASG(gb),     ASL(lb),     0, 0);
    __builtin_amdgcn_global_load_async_to_lds_b128(ASG(gb + 8), ASL(lb + 8), 0, 0);
#else
    v8bf a0 = *(const v8bf*)ga; v8bf a1 = *(const v8bf*)(ga + 8);
    v8bf b0 = *(const v8bf*)gb; v8bf b1 = *(const v8bf*)(gb + 8);
    *(v8bf*)la = a0; *(v8bf*)(la + 8) = a1;
    *(v8bf*)lb = b0; *(v8bf*)(lb + 8) = b1;
#endif
  };

  v8f acc[2][4] = {};

  const int nk = K / BK;
  stage(0, 0);
  for (int kt = 0; kt < nk; ++kt) {
    const int cur = kt & 1;
    if (kt + 1 < nk) {
      stage(cur ^ 1, (kt + 1) * BK);
#if USE_ASYNC_LDS
      __builtin_amdgcn_s_wait_asynccnt(4);   // current buffer's 4 are done
#endif
    } else {
#if USE_ASYNC_LDS
      __builtin_amdgcn_s_wait_asynccnt(0);
#endif
    }
    __syncthreads();

    const bf16* As = &sh[cur][0];
    const bf16* Bs = &sh[cur][BOFF];
    v16bf a[2];
#pragma unroll
    for (int am = 0; am < 2; ++am) {
      const bf16* ap = As + (m0w + am * 16 + lr) * LROW + aK;
      v8bf lo = *(const v8bf*)ap;
      v8bf hi = *(const v8bf*)(ap + 16);
      a[am] = __builtin_shufflevector(lo, hi, 0,1,2,3,4,5,6,7,
                                              8,9,10,11,12,13,14,15);
    }
    v16bf b[4];
#pragma unroll
    for (int bn = 0; bn < 4; ++bn) {
      const bf16* bp = Bs + (n0w + bn * 16 + lr) * LROW + bK;
      v8bf lo = *(const v8bf*)bp;
      v8bf hi = *(const v8bf*)(bp + 16);
      b[bn] = __builtin_shufflevector(lo, hi, 0,1,2,3,4,5,6,7,
                                              8,9,10,11,12,13,14,15);
    }
#pragma unroll
    for (int am = 0; am < 2; ++am)
#pragma unroll
      for (int bn = 0; bn < 4; ++bn)
        acc[am][bn] = __builtin_amdgcn_wmma_f32_16x16x32_bf16(
            false, a[am], false, b[bn], (short)0, acc[am][bn], false, false);

    __syncthreads();   // all waves done reading before buffer is overwritten
  }

  // C/D layout: VGPR r, lanes 0-15: M=r, lanes 16-31: M=r+8; N = lane&15.
#pragma unroll
  for (int am = 0; am < 2; ++am)
#pragma unroll
    for (int bn = 0; bn < 4; ++bn)
#pragma unroll
      for (int r = 0; r < 8; ++r) {
        const int m = bm0 + m0w + am * 16 + lh * 8 + r;
        const int n = bn0 + n0w + bn * 16 + lr;
        const float v = acc[am][bn][r];
        if constexpr (EPI == 0) {
          bf16* C = (bf16*)Cv + (long long)z * sC;
          C[(long long)m * ldc + n] = tobf(v);
        } else if constexpr (EPI == 1) {         // v^T: [b, n, t] <- [b*T+t, n]
          bf16* C = (bf16*)Cv;
          const int bb = m >> 10, t = m & (Tn - 1);
          C[((long long)bb * ldc + n) * Tn + t] = tobf(v);
        } else if constexpr (EPI == 2) {         // f = relu(mask(s*sc))*decay
          float* C = (float*)Cv + (long long)z * sC;
          float o = 0.f;
          if (n <= m) {
            const float decay = 1.0f - 0.1f * (float)(m - n) / (float)Tn;
            o = fmaxf(v * scale, 0.f) * decay;
          }
          C[(long long)m * ldc + n] = o;
        } else if constexpr (EPI == 3) {
          bf16* C = (bf16*)Cv + (long long)z * sC;
          C[(long long)m * ldc + n] = tobf(fmaxf(v, 0.f));
        } else {
          float* C = (float*)Cv + (long long)z * sC;
          C[(long long)m * ldc + n] = v;
        }
      }
}

// One wave per row. MODE 0: probs + per-row entropy; MODE 1: plain softmax.
template <int MODE, int GUARD>
__global__ __launch_bounds__(256)
void softmax_rows(const float* __restrict__ X, bf16* __restrict__ P,
                  float* __restrict__ ent, const int* __restrict__ flag,
                  int ncols)
{
  if constexpr (GUARD == 1) { if (*flag == 0) return; }
  const int lane = threadIdx.x & 31;
  const long long row = (long long)blockIdx.x * 8 + (threadIdx.x >> 5);
  const float* x = X + row * ncols;
  bf16* p = P + row * ncols;

  float mx = -__builtin_inff();
  for (int c = lane; c < ncols; c += 32) mx = fmaxf(mx, x[c]);
#pragma unroll
  for (int o = 16; o > 0; o >>= 1) mx = fmaxf(mx, __shfl_xor(mx, o, 32));

  float s = 0.f;
  for (int c = lane; c < ncols; c += 32) s += expf(x[c] - mx);
#pragma unroll
  for (int o = 16; o > 0; o >>= 1) s += __shfl_xor(s, o, 32);
  const float inv = 1.0f / s;

  float e = 0.f;
  for (int c = lane; c < ncols; c += 32) {
    const float pv = expf(x[c] - mx) * inv;
    p[c] = tobf(pv);
    if constexpr (MODE == 0) e -= pv * logf(pv + 1e-9f);
  }
  if constexpr (MODE == 0) {
#pragma unroll
    for (int o = 16; o > 0; o >>= 1) e += __shfl_xor(e, o, 32);
    if (lane == 0) ent[row] = e;
  }
}

// Deterministic entropy reduction + branch flag + quant params.
__global__ __launch_bounds__(256)
void entropy_flag(const float* __restrict__ ent, int n,
                  int* __restrict__ flag, float* __restrict__ kfamr)
{
  __shared__ float shm[256];
  float s = 0.f;
  for (int i = threadIdx.x; i < n; i += 256) s += ent[i];
  shm[threadIdx.x] = s;
  __syncthreads();
  for (int o = 128; o > 0; o >>= 1) {
    if (threadIdx.x < o) shm[threadIdx.x] += shm[threadIdx.x + o];
    __syncthreads();
  }
  if (threadIdx.x == 0) {
    const float avg = shm[0] / (float)n / logf((float)Tn);
    flag[0] = (avg > 0.5f) ? 1 : 0;
    kfamr[0] = fminf(fmaxf(floorf(4.0f + 12.0f * avg), 4.0f), 16.0f);  // kf
    kfamr[1] = floorf(15.0f + 112.0f * avg);                           // amr
  }
}

// Quant branch (flag==0): one thread per row; top-16 with jax top_k tie order.
__global__ __launch_bounds__(256)
void quant_rows(const float* __restrict__ F, bf16* __restrict__ P,
                const int* __restrict__ flag, const float* __restrict__ kfamr,
                const float* __restrict__ gamma, int ncols)
{
  if (*flag != 0) return;
  const long long row = (long long)blockIdx.x * 256 + threadIdx.x;
  const float* x = F + row * ncols;
  bf16* p = P + row * ncols;

  float s = 0.f, ss = 0.f, mx = -__builtin_inff();
  for (int c = 0; c < ncols; ++c) {
    const float v = x[c];
    s += v; ss += v * v; mx = fmaxf(mx, v);
  }
  const float mean = s / (float)ncols;
  const float var  = fmaxf((ss - (float)ncols * mean * mean) /
                           (float)(ncols - 1), 0.f);
  const float denom = fmaxf(mx, sqrtf(var)) + 1e-6f;
  const float kf = kfamr[0], amr = kfamr[1];
  const float g = gamma[0];

  const bf16 zb = tobf(0.f);
  for (int c = 0; c < ncols; ++c) p[c] = zb;

  float prevV = __builtin_inff(); int prevI = -1;
  for (int r = 0; r < 16; ++r) {
    float best = -__builtin_inff(); int bi = 0;
    for (int c = 0; c < ncols; ++c) {
      const float v = x[c];
      const bool taken = (v > prevV) || (v == prevV && c <= prevI);
      if (!taken && v > best) { best = v; bi = c; }
    }
    float qv = 0.f;
    if ((float)r < kf)
      qv = fminf(fmaxf(floorf(amr * best / denom), 0.f), amr);
    p[bi] = tobf(qv / g);
    prevV = best; prevI = bi;
  }
}

extern "C" void kernel_launch(void* const* d_in, const int* in_sizes, int n_in,
                              void* d_out, int out_size, void* d_ws,
                              size_t ws_size, hipStream_t stream)
{
  (void)in_sizes; (void)n_in; (void)out_size; (void)ws_size;
  const float* x   = (const float*)d_in[0];
  const float* Wq  = (const float*)d_in[1];
  const float* Wk  = (const float*)d_in[2];
  const float* Wv  = (const float*)d_in[3];
  const float* W1  = (const float*)d_in[4];
  const float* W2  = (const float*)d_in[5];
  const float* gam = (const float*)d_in[6];
  float* out = (float*)d_out;

  size_t off = 0;
  auto take = [&](size_t bytes) -> char* {
    char* p = (char*)d_ws + off;
    off += (bytes + 255) & ~(size_t)255;
    return p;
  };
  bf16*  xb   = (bf16*)take((size_t)BTn * Dn * 2);
  bf16*  wqb  = (bf16*)take((size_t)HSn * Dn * 2);
  bf16*  wkb  = (bf16*)take((size_t)HSn * Dn * 2);
  bf16*  wvb  = (bf16*)take((size_t)HSn * Dn * 2);
  bf16*  w1b  = (bf16*)take((size_t)Tn * Ln * 2);
  bf16*  w2b  = (bf16*)take((size_t)Ln * Tn * 2);
  bf16*  qb   = (bf16*)take((size_t)BTn * HSn * 2);
  bf16*  kb   = (bf16*)take((size_t)BTn * HSn * 2);
  bf16*  vtb  = (bf16*)take((size_t)Bn * HSn * Tn * 2);
  float* f    = (float*)take((size_t)BTn * Tn * 4);  // scores f; reused as w
  bf16*  prb  = (bf16*)take((size_t)BTn * Tn * 2);   // probs
  bf16*  ltb  = (bf16*)take((size_t)BTn * Ln * 2);   // latent
  bf16*  atb  = (bf16*)take((size_t)BTn * Tn * 2);   // attn
  float* ent  = (float*)take((size_t)BTn * 4);
  int*   flag = (int*)take(256);
  float* kfamr = (float*)take(256);

  auto cvt = [&](const float* src, bf16* dst, int n) {
    cvt_bf16<<<(n + 255) / 256, 256, 0, stream>>>(src, dst, n);
  };
  cvt(x,  xb,  BTn * Dn);
  cvt(Wq, wqb, HSn * Dn);
  cvt(Wk, wkb, HSn * Dn);
  cvt(Wv, wvb, HSn * Dn);
  cvt(W1, w1b, Tn * Ln);
  cvt(W2, w2b, Ln * Tn);

  // q = x Wq^T ; k = x Wk^T ; v^T from x Wv^T
  gemm_bf16_nt<0, 0><<<dim3(HSn / BN, BTn / BM, 1), 256, 0, stream>>>(
      xb, 0, Dn, wqb, 0, Dn, qb, 0, HSn, Dn, 1.f, flag);
  gemm_bf16_nt<0, 0><<<dim3(HSn / BN, BTn / BM, 1), 256, 0, stream>>>(
      xb, 0, Dn, wkb, 0, Dn, kb, 0, HSn, Dn, 1.f, flag);
  gemm_bf16_nt<1, 0><<<dim3(HSn / BN, BTn / BM, 1), 256, 0, stream>>>(
      xb, 0, Dn, wvb, 0, Dn, vtb, 0, HSn, Dn, 1.f, flag);

  // f[b] = relu(mask(q k^T / sqrt(HS))) * decay   (batched over B)
  const float sc = 1.0f / sqrtf((float)HSn);
  gemm_bf16_nt<2, 0><<<dim3(Tn / BN, Tn / BM, Bn), 256, 0, stream>>>(
      qb, (long long)Tn * HSn, HSn, kb, (long long)Tn * HSn, HSn,
      f, (long long)Tn * Tn, Tn, HSn, sc, flag);

  softmax_rows<0, 0><<<BTn / 8, 256, 0, stream>>>(f, prb, ent, flag, Tn);
  entropy_flag<<<1, 256, 0, stream>>>(ent, BTn, flag, kfamr);

  // latent branch (flag==1): lat = relu(probs W1^T); w = lat W2^T; softmax
  gemm_bf16_nt<3, 1><<<dim3(Tn / BN, BTn / BM, 1), 256, 0, stream>>>(
      prb, 0, Tn, w1b, 0, Ln, ltb, 0, Tn, Tn, 1.f, flag);
  gemm_bf16_nt<4, 1><<<dim3(Ln / BN, BTn / BM, 1), 256, 0, stream>>>(
      ltb, 0, Tn, w2b, 0, Tn, f, 0, Ln, Tn, 1.f, flag);
  softmax_rows<1, 1><<<BTn / 8, 256, 0, stream>>>(f, atb, nullptr, flag, Tn);

  // quant branch (flag==0): top-16 quantization of f -> attn
  quant_rows<<<BTn / 256, 256, 0, stream>>>(f, atb, flag, kfamr, gam, Tn);

  // out[b] = attn[b] @ v[b]  (NT via v^T), fp32 to d_out
  gemm_bf16_nt<4, 0><<<dim3(HSn / BN, Tn / BM, Bn), 256, 0, stream>>>(
      atb, (long long)Tn * Tn, Tn, vtb, (long long)HSn * Tn, Tn,
      out, (long long)Tn * HSn, HSn, Tn, 1.f, flag);
}
